// AttentionModel_43542378447496
// MI455X (gfx1250) — matrix operand order repaired and checked by reading, not compiled, not run on hardware
//
#include <hip/hip_runtime.h>
#include <hip/hip_bf16.h>
#include <math.h>

typedef __attribute__((ext_vector_type(16))) _Float16 v16h;
typedef __attribute__((ext_vector_type(8)))  _Float16 v8h;
typedef __attribute__((ext_vector_type(8)))  float    v8f;

#define BB   1024
#define NN   100
#define EE   128
#define HH   8
#define LL   3
#define FFD  512
#define BN   (BB*NN)          // 102400
#define BNE  (BN*EE)          // 13,107,200

// ---------------------------------------------------------------------------
// Tiled WMMA GEMM:  C(MxN) = A(MxK,f16) * Bt(NxK,f16)^T  [+bias][+resid][relu]
// Block: 256 threads = 8 waves (4 in M x 2 in N); wave tile 32x64 = 2x4 wmma
// tiles of v_wmma_f32_16x16x32_f16; block tile 128x128.
// Per wave k-step: 12 x b128 loads feed 8 WMMAs (0.75KB/WMMA operand traffic).
// ---------------------------------------------------------------------------
__global__ __launch_bounds__(256) void gemm_wmma_kernel(
    const _Float16* __restrict__ A, const _Float16* __restrict__ Bt,
    const float* __restrict__ bias, const float* __restrict__ resid,
    float* __restrict__ outF, _Float16* __restrict__ outH,
    int M, int N, int K, int relu)
{
  const int tid  = threadIdx.x;
  const int lane = tid & 31;
  const int w    = tid >> 5;
  const int wm   = w & 3;
  const int wn   = w >> 2;
  const int row0 = blockIdx.x * 128 + wm * 32;
  const int col0 = blockIdx.y * 128 + wn * 64;

  // A 16x32 operand, contiguous k order (same as B): lane m = lane%16; K run of 16 at (lane/16)*16
  const int mA  = lane & 15;
  const int kbA = (lane >> 4) * 16;
  // B 32x16 operand, contiguous k order (same as A): lane n = lane%16; K run of 16 at (lane/16)*16
  const int nB  = lane & 15;
  const int kbB = (lane >> 4) * 16;

  const _Float16* A0 = A  + (size_t)(row0 + mA) * K + kbA;
  const _Float16* A1 = A0 + (size_t)16 * K;
  const _Float16* B0 = Bt + (size_t)(col0 + nB) * K + kbB;
  const _Float16* B1 = B0 + (size_t)16 * K;
  const _Float16* B2 = B0 + (size_t)32 * K;
  const _Float16* B3 = B0 + (size_t)48 * K;

  v8f z = {0.f,0.f,0.f,0.f,0.f,0.f,0.f,0.f};
  v8f c00 = z, c01 = z, c02 = z, c03 = z;
  v8f c10 = z, c11 = z, c12 = z, c13 = z;

  for (int k0 = 0; k0 < K; k0 += 32) {
    v8h a0l = *(const v8h*)(A0 + k0);
    v8h a0h = *(const v8h*)(A0 + k0 + 8);
    v8h a1l = *(const v8h*)(A1 + k0);
    v8h a1h = *(const v8h*)(A1 + k0 + 8);
    v8h b0l = *(const v8h*)(B0 + k0);
    v8h b0h = *(const v8h*)(B0 + k0 + 8);
    v8h b1l = *(const v8h*)(B1 + k0);
    v8h b1h = *(const v8h*)(B1 + k0 + 8);
    v8h b2l = *(const v8h*)(B2 + k0);
    v8h b2h = *(const v8h*)(B2 + k0 + 8);
    v8h b3l = *(const v8h*)(B3 + k0);
    v8h b3h = *(const v8h*)(B3 + k0 + 8);

    v16h a0 = __builtin_shufflevector(a0l, a0h, 0,1,2,3,4,5,6,7,8,9,10,11,12,13,14,15);
    v16h a1 = __builtin_shufflevector(a1l, a1h, 0,1,2,3,4,5,6,7,8,9,10,11,12,13,14,15);
    v16h b0 = __builtin_shufflevector(b0l, b0h, 0,1,2,3,4,5,6,7,8,9,10,11,12,13,14,15);
    v16h b1 = __builtin_shufflevector(b1l, b1h, 0,1,2,3,4,5,6,7,8,9,10,11,12,13,14,15);
    v16h b2 = __builtin_shufflevector(b2l, b2h, 0,1,2,3,4,5,6,7,8,9,10,11,12,13,14,15);
    v16h b3 = __builtin_shufflevector(b3l, b3h, 0,1,2,3,4,5,6,7,8,9,10,11,12,13,14,15);

    c00 = __builtin_amdgcn_wmma_f32_16x16x32_f16(false, a0, false, b0, (short)0, c00, false, false);
    c01 = __builtin_amdgcn_wmma_f32_16x16x32_f16(false, a0, false, b1, (short)0, c01, false, false);
    c02 = __builtin_amdgcn_wmma_f32_16x16x32_f16(false, a0, false, b2, (short)0, c02, false, false);
    c03 = __builtin_amdgcn_wmma_f32_16x16x32_f16(false, a0, false, b3, (short)0, c03, false, false);
    c10 = __builtin_amdgcn_wmma_f32_16x16x32_f16(false, a1, false, b0, (short)0, c10, false, false);
    c11 = __builtin_amdgcn_wmma_f32_16x16x32_f16(false, a1, false, b1, (short)0, c11, false, false);
    c12 = __builtin_amdgcn_wmma_f32_16x16x32_f16(false, a1, false, b2, (short)0, c12, false, false);
    c13 = __builtin_amdgcn_wmma_f32_16x16x32_f16(false, a1, false, b3, (short)0, c13, false, false);
  }

  // C layout: VGPR r -> row = r + 8*(lane/16); col = lane%16
  const int cn = lane & 15;
  const int rb = (lane >> 4) * 8;
  v8f accs[8] = {c00, c01, c02, c03, c10, c11, c12, c13};
#pragma unroll
  for (int t = 0; t < 8; t++) {
    const int i = t >> 2, j = t & 3;
    const int col = col0 + 16 * j + cn;
    const float bv = bias ? bias[col] : 0.0f;
#pragma unroll
    for (int r = 0; r < 8; r++) {
      const int row = row0 + 16 * i + rb + r;
      const size_t idx = (size_t)row * N + col;
      float v = accs[t][r] + bv;
      if (resid) v += resid[idx];
      if (relu)  v = v > 0.0f ? v : 0.0f;
      if (outF)  outF[idx] = v;
      if (outH)  outH[idx] = (_Float16)v;
    }
  }
}

// ---------------------------------------------------------------------------
// Weight repack: f32 (RxC) -> f16 transposed (CxR)
// ---------------------------------------------------------------------------
__global__ void trans_cvt_kernel(const float* __restrict__ src, _Float16* __restrict__ dst,
                                 int R, int C)
{
  int idx = blockIdx.x * 256 + threadIdx.x;
  if (idx >= R * C) return;
  int r = idx / C, c = idx % C;
  dst[(size_t)c * R + r] = (_Float16)src[idx];
}

// QKV combined repack: dst[l][r][k], r = m*128 + h*16 + kk (m: 0=Q,1=K,2=V), k = e
__global__ void qkv_pack_kernel(const float* __restrict__ Wq, const float* __restrict__ Wk,
                                const float* __restrict__ Wv, _Float16* __restrict__ dst)
{
  int idx = blockIdx.x * 256 + threadIdx.x;   // total 3*384*128
  if (idx >= LL * 384 * EE) return;
  int l   = idx / (384 * EE);
  int rem = idx % (384 * EE);
  int r   = rem / EE;
  int k   = rem % EE;
  int m   = r >> 7;
  int hh  = (r & 127) >> 4;
  int kk  = r & 15;
  const float* src = (m == 0) ? Wq : ((m == 1) ? Wk : Wv);
  dst[idx] = (_Float16)src[(((size_t)l * HH + hh) * EE + k) * 16 + kk];
}

// ---------------------------------------------------------------------------
// h = loc @ init_W + init_b  (K=2, scalar)
// ---------------------------------------------------------------------------
__global__ void init_embed_kernel(const float* __restrict__ loc, const float* __restrict__ W,
                                  const float* __restrict__ b, float* __restrict__ h,
                                  _Float16* __restrict__ h16)
{
  int idx = blockIdx.x * 256 + threadIdx.x;
  if (idx >= BNE) return;
  int e  = idx & 127;
  int bn = idx >> 7;
  float v = loc[(size_t)bn * 2] * W[e] + loc[(size_t)bn * 2 + 1] * W[EE + e] + b[e];
  h[idx]   = v;
  h16[idx] = (_Float16)v;
}

// ---------------------------------------------------------------------------
// TDM (Tensor Data Mover) 2D f16 tile load: global -> LDS (async DMA).
// D# bitfields per CDNA5 ISA:
//   group0: count=1 | lds_addr[63:32] | global_addr[120:64] | type=2 [127:126]
//   group1: data_size(2B)=1 @ [17:16]; tensor_dim0 @ [79:48]; tensor_dim1 @ [111:80];
//           tile_dim0 @ [127:112]; tile_dim1 @ [143:128]; dim0_stride @ [207:160]
// ---------------------------------------------------------------------------
#if defined(__gfx1250__) && __has_builtin(__builtin_amdgcn_tensor_load_to_lds)
#define HAVE_TDM 1
typedef unsigned int uint4v __attribute__((ext_vector_type(4)));
typedef int int4v  __attribute__((ext_vector_type(4)));
typedef int int8v  __attribute__((ext_vector_type(8)));

__device__ __forceinline__ void tdm_load_2d_f16(const _Float16* gptr, unsigned lds_addr,
                                                unsigned tile_w, unsigned tile_h,
                                                unsigned row_stride_elems)
{
  unsigned long long ga = (unsigned long long)(uintptr_t)gptr;
  uint4v g0;
  g0.x = 1u;                                                  // count=1, user mode
  g0.y = lds_addr;                                            // LDS byte address
  g0.z = (unsigned)(ga & 0xffffffffu);                        // global_addr[31:0]
  g0.w = (unsigned)((ga >> 32) & 0x01ffffffu) | (2u << 30);   // global_addr[56:32] | type=2
  int8v g1;
  g1[0] = (int)(1u << 16);                                    // data_size = 1 (2 bytes)
  g1[1] = (int)(tile_w << 16);                                // tensor_dim0[15:0]
  g1[2] = (int)(tile_h << 16);                                // tensor_dim1[15:0]
  g1[3] = (int)(tile_w << 16);                                // tile_dim0
  g1[4] = (int)tile_h;                                        // tile_dim1 (tile_dim2=0)
  g1[5] = (int)row_stride_elems;                              // tensor_dim0_stride[31:0]
  g1[6] = 0;
  g1[7] = 0;
  int4v gz = {0, 0, 0, 0};
#if __clang_major__ >= 23
  int8v gz8 = {0, 0, 0, 0, 0, 0, 0, 0};
  __builtin_amdgcn_tensor_load_to_lds(g0, g1, gz, gz, gz8, 0);
#else
  __builtin_amdgcn_tensor_load_to_lds(g0, g1, gz, gz, 0);
#endif
}
#endif

// ---------------------------------------------------------------------------
// Encoder attention: per (b,h) block; K/V tiles (100x16 f16, row stride 384)
// staged to LDS (TDM DMA when available), online softmax over N=100.
// ---------------------------------------------------------------------------
__global__ __launch_bounds__(128) void enc_attn_kernel(const _Float16* __restrict__ qkv,
                                                       _Float16* __restrict__ hd)
{
  const int b = blockIdx.x >> 3;
  const int h = blockIdx.x & 7;
  __shared__ _Float16 Khs[NN][16];
  __shared__ _Float16 Vhs[NN][16];
  const int tid = threadIdx.x;
  const _Float16* base = qkv + (size_t)b * NN * 384 + h * 16;

#if defined(HAVE_TDM)
  if (tid < 32) {  // one wave issues both DMA descriptors
    tdm_load_2d_f16(base + 128, (unsigned)(uintptr_t)&Khs[0][0], 16u, (unsigned)NN, 384u);
    tdm_load_2d_f16(base + 256, (unsigned)(uintptr_t)&Vhs[0][0], 16u, (unsigned)NN, 384u);
    __builtin_amdgcn_s_wait_tensorcnt(0);
  }
#else
  for (int idx = tid; idx < NN * 16; idx += 128) {
    int n = idx >> 4, kk = idx & 15;
    Khs[n][kk] = base[(size_t)n * 384 + 128 + kk];
    Vhs[n][kk] = base[(size_t)n * 384 + 256 + kk];
  }
#endif
  __syncthreads();

  if (tid < NN) {
    float q[16];
#pragma unroll
    for (int kk = 0; kk < 16; kk++) q[kk] = (float)base[(size_t)tid * 384 + kk];
    float Mx = -1e30f, S = 0.0f, O[16];
#pragma unroll
    for (int kk = 0; kk < 16; kk++) O[kk] = 0.0f;
    for (int m = 0; m < NN; m++) {
      float x = 0.0f;
#pragma unroll
      for (int kk = 0; kk < 16; kk++) x += q[kk] * (float)Khs[m][kk];
      x *= 0.25f;                         // 1/sqrt(KD)
      if (x > Mx) {
        float sc = __expf(Mx - x);
        S *= sc;
#pragma unroll
        for (int kk = 0; kk < 16; kk++) O[kk] *= sc;
        Mx = x;
      }
      float ev = __expf(x - Mx);
      S += ev;
#pragma unroll
      for (int kk = 0; kk < 16; kk++) O[kk] += ev * (float)Vhs[m][kk];
    }
    float inv = 1.0f / S;
    _Float16* out = hd + (size_t)(b * NN + tid) * EE + h * 16;
#pragma unroll
    for (int kk = 0; kk < 16; kk++) out[kk] = (_Float16)(O[kk] * inv);
  }
}

// ---------------------------------------------------------------------------
// BatchNorm: atomic reduction of sum/sumsq per feature, then normalize
// ---------------------------------------------------------------------------
__global__ void zero256_kernel(float* __restrict__ p) { p[threadIdx.x] = 0.0f; }

__global__ __launch_bounds__(256) void bn_reduce_kernel(const float* __restrict__ y,
                                                        float* __restrict__ sums,
                                                        float* __restrict__ sumsq)
{
  const int tid = threadIdx.x;
  const int f   = tid & 127;
  float s = 0.0f, q = 0.0f;
  for (int row = blockIdx.x * 2 + (tid >> 7); row < BN; row += gridDim.x * 2) {
    float v = y[(size_t)row * EE + f];
    s += v; q += v * v;
  }
  __shared__ float ls[256], lq[256];
  ls[tid] = s; lq[tid] = q;
  __syncthreads();
  if (tid < 128) {
    atomicAdd(&sums[f],  ls[tid] + ls[tid + 128]);
    atomicAdd(&sumsq[f], lq[tid] + lq[tid + 128]);
  }
}

__global__ void bn_norm_kernel(const float* __restrict__ y, const float* __restrict__ sums,
                               const float* __restrict__ sumsq, const float* __restrict__ g,
                               const float* __restrict__ bta, float* __restrict__ hout,
                               _Float16* __restrict__ h16)
{
  int idx = blockIdx.x * 256 + threadIdx.x;
  if (idx >= BNE) return;
  int f = idx & 127;
  const float invN = 1.0f / (float)BN;
  float m   = sums[f] * invN;
  float var = sumsq[f] * invN - m * m;
  float v = g[f] * (y[idx] - m) * rsqrtf(var + 1e-5f) + bta[f];
  hout[idx] = v;
  h16[idx]  = (_Float16)v;
}

// ---------------------------------------------------------------------------
// Decoder prep: graph mean + gather first/prev rows into step context
// ---------------------------------------------------------------------------
__global__ __launch_bounds__(128) void dec_prep_kernel(const float* __restrict__ h,
                                                       const int* __restrict__ prev_a,
                                                       const int* __restrict__ first_a,
                                                       _Float16* __restrict__ hmean16,
                                                       _Float16* __restrict__ step16)
{
  const int b = blockIdx.x;
  const int e = threadIdx.x;
  const float* hb = h + (size_t)b * NN * EE;
  float s = 0.0f;
  for (int n = 0; n < NN; n++) s += hb[(size_t)n * EE + e];
  hmean16[(size_t)b * EE + e] = (_Float16)(s * (1.0f / (float)NN));
  int fa = first_a[b], pa = prev_a[b];
  step16[(size_t)b * 256 + e]       = (_Float16)hb[(size_t)fa * EE + e];  // [ef, ep]
  step16[(size_t)b * 256 + 128 + e] = (_Float16)hb[(size_t)pa * EE + e];
}

// ---------------------------------------------------------------------------
// Decoder: masked 8-head glimpse attention + proj_out + clipped logits
// ---------------------------------------------------------------------------
__global__ __launch_bounds__(128) void dec_attn_kernel(const float* __restrict__ query,
                                                       const _Float16* __restrict__ gkvl,
                                                       const _Float16* __restrict__ po_t,
                                                       const int* __restrict__ prev_a,
                                                       const int* __restrict__ first_a,
                                                       float* __restrict__ out)
{
  const int b = blockIdx.x, tid = threadIdx.x;
  __shared__ float q[EE], heads[EE], glimpse[EE], cs[NN], red[2];
  q[tid] = query[(size_t)b * EE + tid];
  const int pa = prev_a[b], fa = first_a[b];
  __syncthreads();
  const _Float16* gb = gkvl + (size_t)b * NN * 384;

  for (int h = 0; h < HH; h++) {
    if (tid < NN) {
      float x = 0.0f;
      const _Float16* kb = gb + (size_t)tid * 384 + h * 16;
#pragma unroll
      for (int kk = 0; kk < 16; kk++) x += q[h * 16 + kk] * (float)kb[kk];
      x *= 0.25f;
      if (tid == pa || tid == fa) x = -INFINITY;
      cs[tid] = x;
    }
    __syncthreads();
    if (tid == 0) {
      float mx = -INFINITY;
      for (int n = 0; n < NN; n++) mx = fmaxf(mx, cs[n]);
      red[0] = mx;
    }
    __syncthreads();
    if (tid < NN) cs[tid] = __expf(cs[tid] - red[0]);
    __syncthreads();
    if (tid == 0) {
      float s = 0.0f;
      for (int n = 0; n < NN; n++) s += cs[n];
      red[1] = s;
    }
    __syncthreads();
    if (tid < 16) {
      float o = 0.0f;
      const _Float16* vb = gb + 128 + h * 16 + tid;
      for (int n = 0; n < NN; n++) o += cs[n] * (float)vb[(size_t)n * 384];
      heads[h * 16 + tid] = o / red[1];
    }
    __syncthreads();
  }
  {
    float gacc = 0.0f;
    const _Float16* pr = po_t + (size_t)tid * EE;  // po_t[e][k]
    for (int k = 0; k < EE; k++) gacc += heads[k] * (float)pr[k];
    glimpse[tid] = gacc;
  }
  __syncthreads();
  if (tid < NN) {
    const _Float16* lk = gb + (size_t)tid * 384 + 256;
    float s = 0.0f;
    for (int e = 0; e < EE; e++) s += glimpse[e] * (float)lk[e];
    s *= 0.08838834764831845f;  // 1/sqrt(E)
    float v = tanhf(s) * 10.0f;
    if (tid == pa || tid == fa) v = -INFINITY;
    out[(size_t)b * NN + tid] = v;
  }
}

// ---------------------------------------------------------------------------
// Host orchestration
// ---------------------------------------------------------------------------
static inline char* ws_take(char* base, size_t& off, size_t bytes)
{
  char* p = base + off;
  off += (bytes + 255) & ~(size_t)255;
  return p;
}

static void gemm(const _Float16* A, const _Float16* Bt, const float* bias,
                 const float* resid, float* outF, _Float16* outH,
                 int M, int N, int K, int relu, hipStream_t s)
{
  dim3 g(M / 128, N / 128), blk(256);
  gemm_wmma_kernel<<<g, blk, 0, s>>>(A, Bt, bias, resid, outF, outH, M, N, K, relu);
}

extern "C" void kernel_launch(void* const* d_in, const int* in_sizes, int n_in,
                              void* d_out, int out_size, void* d_ws, size_t ws_size,
                              hipStream_t stream)
{
  (void)in_sizes; (void)n_in; (void)out_size; (void)ws_size;
  const float* loc     = (const float*)d_in[0];
  const int*   prev_a  = (const int*)d_in[1];
  const int*   first_a = (const int*)d_in[2];
  // d_in[3] = action_mask (recomputed from prev/first instead)
  const float* init_W  = (const float*)d_in[4];
  const float* init_b  = (const float*)d_in[5];
  const float* Wq      = (const float*)d_in[6];
  const float* Wk      = (const float*)d_in[7];
  const float* Wv      = (const float*)d_in[8];
  const float* Wo      = (const float*)d_in[9];
  const float* n1_g    = (const float*)d_in[10];
  const float* n1_b    = (const float*)d_in[11];
  const float* ff1_W   = (const float*)d_in[12];
  const float* ff1_b   = (const float*)d_in[13];
  const float* ff2_W   = (const float*)d_in[14];
  const float* ff2_b   = (const float*)d_in[15];
  const float* n2_g    = (const float*)d_in[16];
  const float* n2_b    = (const float*)d_in[17];
  const float* pnodes  = (const float*)d_in[18];
  const float* pctx    = (const float*)d_in[19];
  const float* pstep   = (const float*)d_in[20];
  const float* pout    = (const float*)d_in[21];

  char* ws = (char*)d_ws;
  size_t off = 0;
  float*     h      = (float*)    ws_take(ws, off, (size_t)BNE * 4);
  _Float16*  h16    = (_Float16*) ws_take(ws, off, (size_t)BNE * 2);
  _Float16*  big16  = (_Float16*) ws_take(ws, off, (size_t)BN * FFD * 2); // qkv / ff tmp / gkvl
  _Float16*  hd16   = (_Float16*) ws_take(ws, off, (size_t)BNE * 2);
  float*     ybuf   = (float*)    ws_take(ws, off, (size_t)BNE * 4);
  float*     stats  = (float*)    ws_take(ws, off, 256 * 4);             // sums[128]|sumsq[128]
  _Float16*  hmean16= (_Float16*) ws_take(ws, off, (size_t)BB * EE * 2);
  _Float16*  step16 = (_Float16*) ws_take(ws, off, (size_t)BB * 256 * 2);
  float*     fctx   = (float*)    ws_take(ws, off, (size_t)BB * EE * 4);
  float*     query  = (float*)    ws_take(ws, off, (size_t)BB * EE * 4);
  _Float16*  qkvw   = (_Float16*) ws_take(ws, off, (size_t)LL * 384 * EE * 2);
  _Float16*  wo_t   = (_Float16*) ws_take(ws, off, (size_t)LL * EE * EE * 2);
  _Float16*  ff1_t  = (_Float16*) ws_take(ws, off, (size_t)LL * EE * FFD * 2);
  _Float16*  ff2_t  = (_Float16*) ws_take(ws, off, (size_t)LL * EE * FFD * 2);
  _Float16*  pn_t   = (_Float16*) ws_take(ws, off, (size_t)384 * EE * 2);
  _Float16*  pc_t   = (_Float16*) ws_take(ws, off, (size_t)EE * EE * 2);
  _Float16*  ps_t   = (_Float16*) ws_take(ws, off, (size_t)EE * 256 * 2);
  _Float16*  po_t   = (_Float16*) ws_take(ws, off, (size_t)EE * EE * 2);

  // ---- weight repack (f32 -> f16 transposed) ----
  qkv_pack_kernel<<<(LL * 384 * EE + 255) / 256, 256, 0, stream>>>(Wq, Wk, Wv, qkvw);
  for (int l = 0; l < LL; l++) {
    trans_cvt_kernel<<<(EE * EE + 255) / 256, 256, 0, stream>>>(Wo + (size_t)l * EE * EE, wo_t + (size_t)l * EE * EE, EE, EE);
    trans_cvt_kernel<<<(EE * FFD + 255) / 256, 256, 0, stream>>>(ff1_W + (size_t)l * EE * FFD, ff1_t + (size_t)l * EE * FFD, EE, FFD);
    trans_cvt_kernel<<<(EE * FFD + 255) / 256, 256, 0, stream>>>(ff2_W + (size_t)l * EE * FFD, ff2_t + (size_t)l * EE * FFD, FFD, EE);
  }
  trans_cvt_kernel<<<(EE * 384 + 255) / 256, 256, 0, stream>>>(pnodes, pn_t, EE, 384);
  trans_cvt_kernel<<<(EE * EE  + 255) / 256, 256, 0, stream>>>(pctx, pc_t, EE, EE);
  trans_cvt_kernel<<<(256 * EE + 255) / 256, 256, 0, stream>>>(pstep, ps_t, 256, EE);
  trans_cvt_kernel<<<(EE * EE  + 255) / 256, 256, 0, stream>>>(pout, po_t, EE, EE);

  // ---- init embedding ----
  init_embed_kernel<<<(BNE + 255) / 256, 256, 0, stream>>>(loc, init_W, init_b, h, h16);

  // ---- encoder layers ----
  for (int l = 0; l < LL; l++) {
    // QKV projection: (BN x 128) @ (128 x 384)
    gemm(h16, qkvw + (size_t)l * 384 * EE, nullptr, nullptr, nullptr, big16, BN, 384, EE, 0, stream);
    enc_attn_kernel<<<BB * HH, 128, 0, stream>>>(big16, hd16);
    // out-proj + residual
    gemm(hd16, wo_t + (size_t)l * EE * EE, nullptr, h, ybuf, nullptr, BN, EE, EE, 0, stream);
    zero256_kernel<<<1, 256, 0, stream>>>(stats);
    bn_reduce_kernel<<<1024, 256, 0, stream>>>(ybuf, stats, stats + 128);
    bn_norm_kernel<<<(BNE + 255) / 256, 256, 0, stream>>>(ybuf, stats, stats + 128,
        n1_g + (size_t)l * EE, n1_b + (size_t)l * EE, h, h16);
    // FF
    gemm(h16, ff1_t + (size_t)l * EE * FFD, ff1_b + (size_t)l * FFD, nullptr, nullptr, big16, BN, FFD, EE, 1, stream);
    gemm(big16, ff2_t + (size_t)l * EE * FFD, ff2_b + (size_t)l * EE, h, ybuf, nullptr, BN, EE, FFD, 0, stream);
    zero256_kernel<<<1, 256, 0, stream>>>(stats);
    bn_reduce_kernel<<<1024, 256, 0, stream>>>(ybuf, stats, stats + 128);
    bn_norm_kernel<<<(BNE + 255) / 256, 256, 0, stream>>>(ybuf, stats, stats + 128,
        n2_g + (size_t)l * EE, n2_b + (size_t)l * EE, h, h16);
  }

  // ---- decoder ----
  dec_prep_kernel<<<BB, 128, 0, stream>>>(h, prev_a, first_a, hmean16, step16);
  gemm(hmean16, pc_t, nullptr, nullptr, fctx, nullptr, BB, EE, EE, 0, stream);          // fixed context
  gemm(step16, ps_t, nullptr, fctx, query, nullptr, BB, EE, 256, 0, stream);            // query = fctx + step@Wstep
  gemm(h16, pn_t, nullptr, nullptr, nullptr, big16, BN, 384, EE, 0, stream);            // gK|gV|lK
  dec_attn_kernel<<<BB, 128, 0, stream>>>(query, big16, po_t, prev_a, first_a, (float*)d_out);
}
